// LocMotionAppearance_17995912970709
// MI455X (gfx1250) — matrix operand order repaired and checked by reading, not compiled, not run on hardware
//
#include <hip/hip_runtime.h>

#define B_    4
#define H_    384
#define W_    384
#define HW    (H_*W_)
#define NSPF  1200
#define NSEG  (B_*NSPF)     // 4800
#define CF    256
#define CSK   512
#define CCAT  576
#define COUT  256
#define CPB   8             // channels per pooling block
#define SLICES 8
#define PPS   (HW/SLICES)   // 18432 pixels per slice

typedef __attribute__((ext_vector_type(16))) __bf16 v16bf;
typedef __attribute__((ext_vector_type(8)))  __bf16 v8bf;
typedef __attribute__((ext_vector_type(8)))  float  v8f;

__device__ __forceinline__ float bnf(float x, float g, float b, float m, float v) {
  return (x - m) * __frsqrt_rn(v + 1e-5f) * g + b;
}

// Native global FP32 atomic add (global_atomic_add_f32), avoiding any CAS-loop fallback.
__device__ __forceinline__ void gatomAdd(float* p, float v) {
  unsafeAtomicAdd(p, v);
}

__global__ void k_zero(float* p, int n) {
  int i = blockIdx.x * blockDim.x + threadIdx.x;
  if (i < n) p[i] = 0.0f;
}

__global__ void k_f32_to_bf16(const float* __restrict__ s, __bf16* __restrict__ d, int n) {
  int i = blockIdx.x * blockDim.x + threadIdx.x;
  if (i < n) d[i] = (__bf16)s[i];
}

// ---- coord pooling: 1 thread per pixel, global f32 atomics (cheap: 2.9M) ----
__global__ void k_pool_coords(const int* __restrict__ labels,
                              const float* __restrict__ fx, const float* __restrict__ fy,
                              float* counts, float* csum) {
  int id = blockIdx.x * blockDim.x + threadIdx.x;
  if (id >= B_ * HW) return;
  int b = id / HW, p = id % HW;
  int h = p / W_, w = p % W_;
  int lab = b * NSPF + labels[id];
  gatomAdd(&counts[lab], 1.0f);
  float* cs = csum + (size_t)lab * 4;
  gatomAdd(cs + 0, (float)h * (1.0f / (H_ - 1)));
  gatomAdd(cs + 1, (float)w * (1.0f / (W_ - 1)));
  gatomAdd(cs + 2, fx[id]);
  gatomAdd(cs + 3, fy[id]);
}

// ---- feature pooling: LDS-accumulated, streams 604MB once ----
__global__ __launch_bounds__(256) void k_pool_feats(const int* __restrict__ labels,
                                                    const float* __restrict__ feats,
                                                    float* featsum) {
  __shared__ float lacc[CPB * NSPF];   // 38400 B
  int bs = blockIdx.x;
  int slice = bs & (SLICES - 1);
  int cg = (bs / SLICES) & ((CF / CPB) - 1);
  int b  = bs / (SLICES * (CF / CPB));
  for (int j = threadIdx.x; j < CPB * NSPF; j += blockDim.x) lacc[j] = 0.0f;
  __syncthreads();
  int p0 = slice * PPS;
  const int* lb = labels + (size_t)b * HW;
  const float* fb = feats + ((size_t)b * CF + (size_t)cg * CPB) * HW;
  for (int p = p0 + threadIdx.x; p < p0 + PPS; p += blockDim.x) {
    __builtin_prefetch(fb + p + 2048, 0, 0);   // global_prefetch_b8 hint on the stream
    int lab = lb[p];
    #pragma unroll
    for (int c = 0; c < CPB; ++c)
      atomicAdd(&lacc[c * NSPF + lab], fb[(size_t)c * HW + p]);  // ds_add_f32
  }
  __syncthreads();
  for (int j = threadIdx.x; j < CPB * NSPF; j += blockDim.x) {
    int c = j / NSPF, lab = j % NSPF;
    float v = lacc[j];
    if (v != 0.0f)
      gatomAdd(&featsum[(size_t)(b * NSPF + lab) * CF + cg * CPB + c], v);
  }
}

// ---- coords finalize + 4->32->64 MLP (tiny), writes cat[:,0:64] and pos ----
__global__ void k_coords_mlp(const float* __restrict__ counts, const float* __restrict__ csum,
                             const float* bnc_g, const float* bnc_b, const float* bnc_m, const float* bnc_v,
                             const float* __restrict__ c1w, const float* __restrict__ c1b,
                             const float* __restrict__ c2w, const float* __restrict__ c2b,
                             __bf16* cat, float* out_pos) {
  int i = blockIdx.x * blockDim.x + threadIdx.x;
  if (i >= NSEG) return;
  float inv = 1.0f / fmaxf(counts[i], 1.0f);
  float x[4];
  #pragma unroll
  for (int j = 0; j < 4; ++j) x[j] = csum[(size_t)i * 4 + j] * inv;
  out_pos[(size_t)i * 2 + 0] = x[0];
  out_pos[(size_t)i * 2 + 1] = x[1];
  #pragma unroll
  for (int j = 0; j < 4; ++j) x[j] = bnf(x[j], bnc_g[j], bnc_b[j], bnc_m[j], bnc_v[j]);
  float h1[32];
  #pragma unroll
  for (int o = 0; o < 32; ++o) {
    float s = c1b[o];
    #pragma unroll
    for (int j = 0; j < 4; ++j) s += c1w[o * 4 + j] * x[j];
    h1[o] = s;
  }
  __bf16* crow = cat + (size_t)i * CCAT;
  for (int o = 0; o < 64; ++o) {
    float s = c2b[o];
    #pragma unroll
    for (int j = 0; j < 32; ++j) s += c2w[o * 32 + j] * h1[j];
    crow[o] = (__bf16)fmaxf(s, 0.0f);
  }
}

// ---- feature means + BN256 -> bf16 GEMM operand ----
__global__ void k_feats_bn(const float* __restrict__ featsum, const float* __restrict__ counts,
                           const float* g, const float* bb, const float* m, const float* v,
                           __bf16* X) {
  int id = blockIdx.x * blockDim.x + threadIdx.x;
  if (id >= NSEG * CF) return;
  int i = id / CF, c = id % CF;
  float x = featsum[id] / fmaxf(counts[i], 1.0f);
  X[id] = (__bf16)bnf(x, g[c], bb[c], m[c], v[c]);
}

// ---- GEMM1: [4800,256] x [512,256]^T via v_wmma_f32_16x16x32_bf16; fused bias+BN512+ReLU ----
__global__ __launch_bounds__(128) void k_gemm_skip(const __bf16* __restrict__ X, const __bf16* __restrict__ Wt,
                                                   const float* __restrict__ bias,
                                                   const float* g, const float* bb, const float* m, const float* v,
                                                   __bf16* cat) {
  int wave = threadIdx.x >> 5, lane = threadIdx.x & 31;
  int tile = blockIdx.x * 4 + wave;
  int tm = tile >> 5;          // 32 tiles along N=512
  int tn = tile & 31;
  int col16 = lane & 15, half = lane >> 4;
  const __bf16* pa = X  + (size_t)(tm * 16 + col16) * CF + half * 16;
  const __bf16* pb = Wt + (size_t)(tn * 16 + col16) * CF + half * 16;
  v8f acc = {};
  #pragma unroll
  for (int ks = 0; ks < CF / 32; ++ks) {
    union { v16bf v; v8bf h[2]; } a;
    a.h[0] = *(const v8bf*)(pa + ks * 32);
    a.h[1] = *(const v8bf*)(pa + ks * 32 + 8);
    v16bf bfr = *(const v16bf*)(pb + ks * 32);
    acc = __builtin_amdgcn_wmma_f32_16x16x32_bf16(false, a.v, false, bfr, (short)0, acc, false, false);
  }
  int n = tn * 16 + col16;
  float bn_m = m[n], bn_rs = __frsqrt_rn(v[n] + 1e-5f) * g[n], bn_b = bb[n], bi = bias[n];
  #pragma unroll
  for (int i = 0; i < 8; ++i) {
    int row = tm * 16 + half * 8 + i;
    float val = (acc[i] + bi - bn_m) * bn_rs + bn_b;
    cat[(size_t)row * CCAT + 64 + n] = (__bf16)fmaxf(val, 0.0f);
  }
}

// ---- GEMM2: [4800,576] x [256,576]^T -> siam f32 ----
__global__ __launch_bounds__(128) void k_gemm_merge(const __bf16* __restrict__ Xc, const __bf16* __restrict__ Wt,
                                                    const float* __restrict__ bias, float* __restrict__ out) {
  int wave = threadIdx.x >> 5, lane = threadIdx.x & 31;
  int tile = blockIdx.x * 4 + wave;
  int tm = tile >> 4;          // 16 tiles along N=256
  int tn = tile & 15;
  int col16 = lane & 15, half = lane >> 4;
  const __bf16* pa = Xc + (size_t)(tm * 16 + col16) * CCAT + half * 16;
  const __bf16* pb = Wt + (size_t)(tn * 16 + col16) * CCAT + half * 16;
  v8f acc = {};
  #pragma unroll
  for (int ks = 0; ks < CCAT / 32; ++ks) {   // 18 steps
    union { v16bf v; v8bf h[2]; } a;
    a.h[0] = *(const v8bf*)(pa + ks * 32);
    a.h[1] = *(const v8bf*)(pa + ks * 32 + 8);
    v16bf bfr = *(const v16bf*)(pb + ks * 32);
    acc = __builtin_amdgcn_wmma_f32_16x16x32_bf16(false, a.v, false, bfr, (short)0, acc, false, false);
  }
  int n = tn * 16 + col16;
  float bi = bias[n];
  #pragma unroll
  for (int i = 0; i < 8; ++i) {
    int row = tm * 16 + half * 8 + i;
    out[(size_t)row * COUT + n] = acc[i] + bi;
  }
}

extern "C" void kernel_launch(void* const* d_in, const int* in_sizes, int n_in,
                              void* d_out, int out_size, void* d_ws, size_t ws_size,
                              hipStream_t stream) {
  const int*   labels = (const int*)d_in[0];
  const float* fx     = (const float*)d_in[1];
  const float* fy     = (const float*)d_in[2];
  const float* feats  = (const float*)d_in[3];
  const float *bnc_g=(const float*)d_in[4],  *bnc_b=(const float*)d_in[5],
              *bnc_m=(const float*)d_in[6],  *bnc_v=(const float*)d_in[7];
  const float *c1w=(const float*)d_in[8],    *c1b=(const float*)d_in[9],
              *c2w=(const float*)d_in[10],   *c2b=(const float*)d_in[11];
  const float *bnr_g=(const float*)d_in[12], *bnr_b=(const float*)d_in[13],
              *bnr_m=(const float*)d_in[14], *bnr_v=(const float*)d_in[15];
  const float *srw=(const float*)d_in[16],   *srb=(const float*)d_in[17];
  const float *bns_g=(const float*)d_in[18], *bns_b=(const float*)d_in[19],
              *bns_m=(const float*)d_in[20], *bns_v=(const float*)d_in[21];
  const float *mgw=(const float*)d_in[22],   *mgb=(const float*)d_in[23];

  char* ws = (char*)d_ws;
  size_t off = 0;
  auto alloc = [&](size_t bytes) -> void* {
    void* p = ws + off;
    off = (off + bytes + 255) & ~(size_t)255;
    return p;
  };
  float*  counts  = (float*) alloc((size_t)NSEG * 4);
  float*  csum    = (float*) alloc((size_t)NSEG * 4 * 4);
  float*  featsum = (float*) alloc((size_t)NSEG * CF * 4);
  __bf16* Xbf     = (__bf16*)alloc((size_t)NSEG * CF * 2);
  __bf16* cat     = (__bf16*)alloc((size_t)NSEG * CCAT * 2);
  __bf16* srwb    = (__bf16*)alloc((size_t)CSK * CF * 2);
  __bf16* mgwb    = (__bf16*)alloc((size_t)COUT * CCAT * 2);

  // zero the accumulator region (counts..featsum, contiguous incl. padding)
  int zn = (int)((((char*)featsum + (size_t)NSEG * CF * 4) - (char*)counts) / 4);
  k_zero<<<(zn + 255) / 256, 256, 0, stream>>>(counts, zn);

  k_f32_to_bf16<<<(CSK * CF + 255) / 256, 256, 0, stream>>>(srw, srwb, CSK * CF);
  k_f32_to_bf16<<<(COUT * CCAT + 255) / 256, 256, 0, stream>>>(mgw, mgwb, COUT * CCAT);

  k_pool_coords<<<(B_ * HW + 255) / 256, 256, 0, stream>>>(labels, fx, fy, counts, csum);
  k_pool_feats<<<B_ * (CF / CPB) * SLICES, 256, 0, stream>>>(labels, feats, featsum);

  float* siam = (float*)d_out;
  float* pos  = siam + (size_t)NSEG * COUT;
  k_coords_mlp<<<(NSEG + 127) / 128, 128, 0, stream>>>(counts, csum, bnc_g, bnc_b, bnc_m, bnc_v,
                                                       c1w, c1b, c2w, c2b, cat, pos);
  k_feats_bn<<<(NSEG * CF + 255) / 256, 256, 0, stream>>>(featsum, counts, bnr_g, bnr_b, bnr_m, bnr_v, Xbf);

  k_gemm_skip <<<((NSEG / 16) * (CSK / 16)) / 4, 128, 0, stream>>>(Xbf, srwb, srb, bns_g, bns_b, bns_m, bns_v, cat);
  k_gemm_merge<<<((NSEG / 16) * (COUT / 16)) / 4, 128, 0, stream>>>(cat, mgwb, mgb, siam);
}
